// ContextualAttention_27882927686129
// MI455X (gfx1250) — matrix operand order repaired and checked by reading, not compiled, not run on hardware
//
#include <hip/hip_runtime.h>
#include <hip/hip_bf16.h>

typedef __attribute__((ext_vector_type(16))) __bf16 v16bf;
typedef __attribute__((ext_vector_type(8)))  float  v8f;
typedef __attribute__((ext_vector_type(8)))  unsigned short us8;

constexpr int Bn = 2, Cc = 128, Ll = 4096, Dd = 1152;   // H=W=64, L=H*W, D=C*9
constexpr int Mv = 63 * 63;                              // valid agg output rows

__device__ inline unsigned short f2bf(float x) {
    unsigned u = __float_as_uint(x);
    unsigned r = u + 0x7FFFu + ((u >> 16) & 1u);
    return (unsigned short)(r >> 16);
}
__device__ inline int pswap(int l) { return ((l & 63) << 6) | (l >> 6); }

// low 32 bits of a generic pointer to __shared__ == raw LDS byte offset (ISA 10.2)
__device__ inline unsigned lds_addr_of(const void* p) {
    return (unsigned)(size_t)p;
}
// async copy 16B global -> LDS ; tracked by ASYNCcnt
__device__ inline void async_copy16(unsigned lds, const void* gaddr) {
    asm volatile("global_load_async_to_lds_b128 %0, %1, off"
                 :: "v"(lds), "v"(gaddr) : "memory");
}
__device__ inline void async_copy32(unsigned lds, const void* gaddr) {
    asm volatile("global_load_async_to_lds_b128 %0, %1, off\n\t"
                 "global_load_async_to_lds_b128 %0, %1, off offset:16"
                 :: "v"(lds), "v"(gaddr) : "memory");
}
__device__ inline void wait_async0() {
    asm volatile("s_wait_asynccnt 0" ::: "memory");
}

// ---------------------------------------------------------------- utilities
__global__ void zero_out_kernel(float* out, int n) {
    int i = blockIdx.x * blockDim.x + threadIdx.x;
    if (i < n) out[i] = 0.f;
}

__global__ void mm_kernel(const float* mask, float* mm) {
    int k = blockIdx.x * blockDim.x + threadIdx.x;
    if (k >= Ll) return;
    int ky = k >> 6, kx = k & 63;
    float s = 0.f;
    for (int i = 0; i < 3; ++i)
        for (int j = 0; j < 3; ++j) {
            int y = ky + i - 1, x = kx + j - 1;
            if (y >= 0 && y < 64 && x >= 0 && x < 64) s += mask[y * 64 + x];
        }
    mm[k] = (s == 0.f) ? 1.f : 0.f;
}

// ------------------------------------------------- patch matrices (bf16)
__global__ __launch_bounds__(256) void build_patch_kernel(const float* __restrict__ src,
                                                          unsigned short* __restrict__ dst,
                                                          int normalize) {
    int k = blockIdx.x;
    int ky = k >> 6, kx = k & 63;
    int t = threadIdx.x;
    __shared__ float red[256];
    float vals[5];
    int nd = 0;
    float ss = 0.f;
    for (int d = t; d < Dd; d += 256) {
        int c = d / 9, r = d % 9, i = r / 3, j = r % 3;
        int y = ky + i - 1, x = kx + j - 1;
        float v = (y >= 0 && y < 64 && x >= 0 && x < 64) ? src[(c * 64 + y) * 64 + x] : 0.f;
        vals[nd++] = v;
        ss += v * v;
    }
    float inv = 1.f;
    if (normalize) {
        red[t] = ss;
        __syncthreads();
        for (int s = 128; s > 0; s >>= 1) {
            if (t < s) red[t] += red[t + s];
            __syncthreads();
        }
        inv = 1.f / fmaxf(sqrtf(red[0]), 1e-4f);
    }
    nd = 0;
    for (int d = t; d < Dd; d += 256) dst[(size_t)k * Dd + d] = f2bf(vals[nd++] * inv);
}

// rw[(o*C + c)*L + k] = raw_w[k,c,1-i,1-j], o = i*2+j
__global__ void build_rw_kernel(const float* __restrict__ src, unsigned short* __restrict__ rw) {
    int idx = blockIdx.x * blockDim.x + threadIdx.x;
    if (idx >= 4 * Ll * Cc) return;
    int k = idx & 4095, c = (idx >> 12) & 127, o = idx >> 19;
    int di = 1 - (o >> 1), dj = 1 - (o & 1);
    int ky = k >> 6, kx = k & 63;
    int y = ky + di, x = kx + dj;
    float v = (y < 64 && x < 64) ? src[(c * 64 + y) * 64 + x] : 0.f;
    rw[idx] = f2bf(v);
}

// ------------------------------------------------- scores = Wn (LxD) x Fp^T  (fp32 out)
// block: 256 thr = 8 waves ; block tile 128(M) x 128(N) ; K-loop over Dd step 32
__global__ __launch_bounds__(256) void scores_gemm_kernel(const unsigned short* __restrict__ Wn,
                                                          const unsigned short* __restrict__ Fp,
                                                          float* __restrict__ S) {
    __shared__ __align__(32) unsigned short Bs[32 * 128];   // [q][kk] Fp tile
    int t = threadIdx.x, wave = t >> 5, lane = t & 31;
    int m_base = blockIdx.x * 128 + wave * 16;
    int q0 = blockIdx.y * 128;
    const unsigned short* arow = Wn + (size_t)(m_base + (lane & 15)) * Dd + ((lane >> 4) << 4);

    v8f acc[8];
#pragma unroll
    for (int j = 0; j < 8; ++j) acc[j] = v8f{0.f, 0.f, 0.f, 0.f, 0.f, 0.f, 0.f, 0.f};

    int qi = t >> 1, kq = (t & 1) << 4;       // 128 rows x 2 k-halves of 16
    for (int k0 = 0; k0 < Dd; k0 += 32) {
        const unsigned short* p = Fp + (size_t)(q0 + qi) * Dd + k0 + kq;
        us8 tmp0 = *reinterpret_cast<const us8*>(p);
        us8 tmp1 = *reinterpret_cast<const us8*>(p + 8);
        __syncthreads();                        // previous iter's LDS reads complete
#pragma unroll
        for (int e = 0; e < 8; ++e) {
            Bs[qi * 32 + kq + e]     = tmp0[e];
            Bs[qi * 32 + kq + e + 8] = tmp1[e];
        }
        __syncthreads();

        v16bf a = *reinterpret_cast<const v16bf*>(arow + k0);
#pragma unroll
        for (int j = 0; j < 8; ++j) {
            v16bf bmat = *reinterpret_cast<const v16bf*>(&Bs[(j * 16 + (lane & 15)) * 32 + ((lane >> 4) << 4)]);
            acc[j] = __builtin_amdgcn_wmma_f32_16x16x32_bf16(false, a, false, bmat,
                                                             (short)0, acc[j], false, false);
        }
    }
#pragma unroll
    for (int j = 0; j < 8; ++j) {
        int n = q0 + j * 16 + (lane & 15);
#pragma unroll
        for (int r = 0; r < 8; ++r) {
            int m = m_base + r + ((lane >> 4) << 3);
            S[(size_t)m * Ll + n] = acc[j][r];
        }
    }
}

// ------------------------------------------------- fuse (two diagonal 3-tap passes)
__global__ void fuse1_kernel(const float* __restrict__ S, float* __restrict__ F1) {
    size_t idx = (size_t)blockIdx.x * blockDim.x + threadIdx.x;
    if (idx >= (size_t)Ll * Ll) return;
    int l1 = (int)(idx / Ll), l2 = (int)(idx % Ll);
    float s = 0.f;
#pragma unroll
    for (int tt = -1; tt <= 1; ++tt) {
        int a = l1 + tt, b = l2 + tt;
        if (a >= 0 && a < Ll && b >= 0 && b < Ll) s += S[(size_t)a * Ll + b];
    }
    F1[idx] = s;
}

__global__ void fuse2_kernel(const float* __restrict__ F1, float* __restrict__ F2) {
    size_t idx = (size_t)blockIdx.x * blockDim.x + threadIdx.x;
    if (idx >= (size_t)Ll * Ll) return;
    int k = (int)(idx / Ll), q = (int)(idx % Ll);
    int pk = pswap(k), pq = pswap(q);
    float s = 0.f;
#pragma unroll
    for (int tt = -1; tt <= 1; ++tt) {
        int a = pk + tt, b = pq + tt;
        if (a >= 0 && a < Ll && b >= 0 && b < Ll)
            s += F1[(size_t)pswap(a) * Ll + pswap(b)];
    }
    F2[idx] = s;
}

// ------------------------------------------------- softmax over keys -> AT[q][k] bf16
__global__ __launch_bounds__(256) void softmax_kernel(const float* __restrict__ S,
                                                      const float* __restrict__ mm,
                                                      unsigned short* __restrict__ AT) {
    __shared__ __align__(16) float smm[Ll];                 // 16 KB
    __shared__ float rM[4][64], rS[4][64];
    int t = threadIdx.x, qi = t & 63, sl = t >> 6;
    int q = blockIdx.x * 64 + qi;

    // async bulk copy mm[] (16KB) global -> LDS: 256 lanes x 16B x 4 beats
    unsigned smmbase = lds_addr_of(smm);
    for (int base = t * 16; base < Ll * 4; base += 256 * 16)
        async_copy16(smmbase + base, (const char*)mm + base);
    wait_async0();
    __syncthreads();

    float mval = -1e30f, sum = 0.f;
    for (int k = sl; k < Ll; k += 4) {
        float z = S[(size_t)k * Ll + q] * smm[k] * 10.f;
        float mn = fmaxf(mval, z);
        sum = sum * __expf(mval - mn) + __expf(z - mn);
        mval = mn;
    }
    rM[sl][qi] = mval;
    rS[sl][qi] = sum;
    __syncthreads();
    if (sl == 0) {
        float M = rM[0][qi], Sm = rS[0][qi];
#pragma unroll
        for (int s2 = 1; s2 < 4; ++s2) {
            float m2 = rM[s2][qi], s3 = rS[s2][qi];
            float mn = fmaxf(M, m2);
            Sm = Sm * __expf(M - mn) + s3 * __expf(m2 - mn);
            M = mn;
        }
        rM[0][qi] = M;
        rS[0][qi] = 1.f / Sm;
    }
    __syncthreads();
    float M = rM[0][qi], inv = rS[0][qi];
    for (int k = sl; k < Ll; k += 4) {
        float z = S[(size_t)k * Ll + q] * smm[k] * 10.f;
        float w = __expf(z - M) * inv * smm[k];
        AT[(size_t)q * Ll + k] = f2bf(w);
    }
}

// ------------------------------------------------- agg: out[q',c] = sum_o AT_o x Rw_o
// block = 8 waves over M ; B tile (32x16 bf16) async-staged in LDS, shared by all waves
__global__ __launch_bounds__(256) void agg_gemm_kernel(const unsigned short* __restrict__ AT,
                                                       const unsigned short* __restrict__ Rw,
                                                       float* __restrict__ out) {
    __shared__ __align__(32) unsigned short BsA[32 * 16];   // [c][kk] row-major, 1 KB
    int t = threadIdx.x, wave = t >> 5, lane = t & 31;
    int m0 = (blockIdx.x * 8 + wave) * 16;
    int n0 = blockIdx.y * 16;
    unsigned bsbase = lds_addr_of(BsA);

    v8f acc = {0.f, 0.f, 0.f, 0.f, 0.f, 0.f, 0.f, 0.f};
    int mrow = m0 + (lane & 15);
#pragma unroll
    for (int o = 0; o < 4; ++o) {
        int i = o >> 1, j = o & 1;
        int qrow = 0;
        if (mrow < Mv) {
            int y = mrow / 63, x = mrow % 63;
            qrow = (y + i) * 64 + (x + j);
        }
        const unsigned short* ap = AT + (size_t)qrow * Ll + ((lane >> 4) << 4);
        const unsigned short* bp = Rw + (size_t)o * Ll * Cc + (size_t)(n0 + (lane >> 1)) * Ll + ((lane & 1) << 4);
        for (int k0 = 0; k0 < Ll; k0 += 32) {
            if (wave == 0) {                    // async stage B cols (lane = col*2 + k-half)
                async_copy32(bsbase + lane * 32, bp + k0);
                wait_async0();
            }
            __syncthreads();
            __builtin_prefetch(ap + k0 + 512, 0, 0);   // global_prefetch next A chunk
            v16bf a  = *reinterpret_cast<const v16bf*>(ap + k0);
            v16bf bm = *reinterpret_cast<const v16bf*>(&BsA[(lane & 15) * 32 + ((lane >> 4) << 4)]);
            acc = __builtin_amdgcn_wmma_f32_16x16x32_bf16(false, a, false, bm,
                                                          (short)0, acc, false, false);
            __syncthreads();
        }
    }
    int c = n0 + (lane & 15);
#pragma unroll
    for (int r = 0; r < 8; ++r) {
        int m = m0 + r + ((lane >> 4) << 3);
        if (m < Mv) {
            int y = m / 63, x = m % 63;
            out[((size_t)c * 64 + y) * 64 + x] = acc[r] * 0.25f;
        }
    }
}

// ---------------------------------------------------------------- launcher
extern "C" void kernel_launch(void* const* d_in, const int* in_sizes, int n_in,
                              void* d_out, int out_size, void* d_ws, size_t ws_size,
                              hipStream_t stream) {
    const float* f    = (const float*)d_in[0];
    const float* bsrc = (const float*)d_in[1];
    const float* mask = (const float*)d_in[2];
    float* out = (float*)d_out;

    char* ws = (char*)d_ws;
    size_t off = 0;
    auto alloc = [&](size_t bytes) {
        size_t p = off;
        off += (bytes + 255) & ~(size_t)255;
        return p;
    };
    unsigned short* Wn   = (unsigned short*)(ws + alloc((size_t)Ll * Dd * 2));
    unsigned short* Fp   = (unsigned short*)(ws + alloc((size_t)Ll * Dd * 2));
    unsigned short* Rw   = (unsigned short*)(ws + alloc((size_t)4 * Ll * Cc * 2));
    float*          mmv  = (float*)         (ws + alloc((size_t)Ll * 4));
    float*          bufS = (float*)         (ws + alloc((size_t)Ll * Ll * 4));
    float*          bufF = (float*)         (ws + alloc((size_t)Ll * Ll * 4));
    unsigned short* AT   = (unsigned short*)bufF;   // reuse bufF for bf16 attn^T

    int outN = Bn * Cc * 64 * 64;
    zero_out_kernel<<<(outN + 255) / 256, 256, 0, stream>>>(out, outN);
    mm_kernel<<<(Ll + 255) / 256, 256, 0, stream>>>(mask, mmv);

    size_t NN = (size_t)Ll * Ll;
    for (int b = 0; b < Bn; ++b) {
        const float* fb = f + (size_t)b * Cc * 64 * 64;
        const float* bb = bsrc + (size_t)b * Cc * 64 * 64;

        build_patch_kernel<<<Ll, 256, 0, stream>>>(bb, Wn, 1);
        build_patch_kernel<<<Ll, 256, 0, stream>>>(fb, Fp, 0);
        build_rw_kernel<<<(4 * Ll * Cc + 255) / 256, 256, 0, stream>>>(bb, Rw);

        dim3 g1(Ll / 128, Ll / 128);
        scores_gemm_kernel<<<g1, 256, 0, stream>>>(Wn, Fp, bufS);

        fuse1_kernel<<<(unsigned)((NN + 255) / 256), 256, 0, stream>>>(bufS, bufF);
        fuse2_kernel<<<(unsigned)((NN + 255) / 256), 256, 0, stream>>>(bufF, bufS);

        softmax_kernel<<<Ll / 64, 256, 0, stream>>>(bufS, mmv, AT);

        dim3 g2(32, 8);
        agg_gemm_kernel<<<g2, 256, 0, stream>>>(AT, Rw, out + (size_t)b * Cc * 64 * 64);
    }
}
